// VectorQuantizer_39462159516041
// MI455X (gfx1250) — matrix operand order repaired and checked by reading, not compiled, not run on hardware
//
#include <hip/hip_runtime.h>

// ---------------- problem constants ----------------
#define B_SZ   16
#define D_CH   256
#define HW     1024          // 32*32
#define NTOK   (B_SZ * HW)   // 16384 tokens
#define NE     8192
#define NTILES (NE / 16)     // 512
#define NSPLIT 8
#define TOK_PER_BLOCK 256    // 8 waves * 32 tokens
#define MBLOCKS (NTOK / TOK_PER_BLOCK)   // 64
#define ZQ_ELEMS  (NTOK * D_CH)          // 4,194,304
#define OUT_IDX_OFF   ZQ_ELEMS
#define OUT_LOSS_OFF  (ZQ_ELEMS + NTOK)
#define OUT_DIV_OFF   (ZQ_ELEMS + NTOK + 1)
#define GATHER_BLOCKS (ZQ_ELEMS / 256)   // 16384
#define BT_STRIDE 264                    // padded LDS row stride (halfs)

typedef unsigned short ushort_t;
typedef __attribute__((ext_vector_type(16))) __bf16 v16bf;
typedef __attribute__((ext_vector_type(8)))  float  v8f;

__device__ __forceinline__ ushort_t f2bf(float f) {
    unsigned u = __float_as_uint(f);
    unsigned r = u + 0x7FFFu + ((u >> 16) & 1u);   // round-to-nearest-even
    return (ushort_t)(r >> 16);
}

// ---------------- kernel 1: zero presence bitmap ----------------
__global__ __launch_bounds__(256) void vq_zero_presence(unsigned* __restrict__ presence) {
    int i = blockIdx.x * 256 + threadIdx.x;        // 512 blocks -> 131072
    presence[i] = 0u;
}

// ---------------- kernel 2: z [B,D,H,W] f32 -> zt [t,d] bf16 ----------------
__global__ __launch_bounds__(256) void vq_prep_z(const float* __restrict__ z,
                                                 ushort_t* __restrict__ zt) {
    int t = blockIdx.x * 256 + threadIdx.x;        // 64 blocks -> 16384 tokens
    int b = t >> 10, hw = t & 1023;
    const float* zp = z + (size_t)b * D_CH * HW + hw;  // coalesced over hw per d
    ushort_t* o = zt + (size_t)t * D_CH;               // 512B-aligned
    for (int d8 = 0; d8 < D_CH; d8 += 8) {
        ushort_t tmp[8];
#pragma unroll
        for (int j = 0; j < 8; ++j) tmp[j] = f2bf(zp[(d8 + j) * HW]);
        *(uint4*)(o + d8) = *(const uint4*)tmp;
    }
}

// ---------------- kernel 3: codebook f32 -> bf16 + sumsq ----------------
__global__ __launch_bounds__(256) void vq_prep_cb(const float* __restrict__ cb,
                                                  ushort_t* __restrict__ cbf,
                                                  float* __restrict__ sumsq) {
    int wave = threadIdx.x >> 5, lane = threadIdx.x & 31;
    int n = blockIdx.x * 8 + wave;                 // 1024 blocks -> 8192 rows
    const float* row = cb + (size_t)n * D_CH;
    int base = lane * 8;
    float s = 0.f;
    ushort_t tmp[8];
#pragma unroll
    for (int j = 0; j < 8; ++j) { float v = row[base + j]; s += v * v; tmp[j] = f2bf(v); }
    *(uint4*)(cbf + (size_t)n * D_CH + base) = *(const uint4*)tmp;
#pragma unroll
    for (int off = 16; off > 0; off >>= 1) s += __shfl_xor(s, off, 32);
    if (lane == 0) sumsq[n] = s;
}

// ---------------- kernel 4: fused bf16-WMMA GEMM + running argmin ----------------
// grid = 64 M-blocks x 8 N-splits = 512 blocks; 8 waves/block; 32 tokens/wave
// (two A-tiles per wave -> each LDS B-fragment feeds TWO independent WMMA chains).
// Global->LDS staging is software-pipelined: register prefetch + double-buffered LDS,
// one barrier per iteration; global latency hidden behind the 16-WMMA burst.
__global__ __launch_bounds__(256) void vq_argmin(const ushort_t* __restrict__ zt,
                                                 const ushort_t* __restrict__ cbf,
                                                 const float*    __restrict__ sumsq,
                                                 float* __restrict__ bestd_ws,   // [NSPLIT][NTOK]
                                                 int*   __restrict__ bestn_ws) { // [NSPLIT][NTOK]
    __shared__ __align__(16) ushort_t Bt[2][16 * BT_STRIDE];  // 2 x 8.25 KiB

    const int tid  = threadIdx.x;
    const int wave = tid >> 5, lane = tid & 31;
    const int hi = lane >> 4, lo = lane & 15;
    const int mblk  = blockIdx.x >> 3;
    const int split = blockIdx.x & 7;
    const int m0 = mblk * TOK_PER_BLOCK + wave * 32;

    // ---- two 16x256 A-tiles fully in registers (ISA 16-bit A layout) ----
    // lane (hi,lo): row M = lo; halves 0..7 = K kt*32+hi*8.. ; halves 8..15 = +16
    v16bf a0[8], a1[8];
    {
        const ushort_t* zr0 = zt + (size_t)(m0 + lo) * D_CH;
        const ushort_t* zr1 = zt + (size_t)(m0 + 16 + lo) * D_CH;
#pragma unroll
        for (int kt = 0; kt < 8; ++kt) {
            int kb = kt * 32 + hi * 8;
            union { uint4 u[2]; v16bf v; } ua;
            ua.u[0] = *(const uint4*)(zr0 + kb);
            ua.u[1] = *(const uint4*)(zr0 + kb + 16);
            a0[kt] = ua.v;
            ua.u[0] = *(const uint4*)(zr1 + kb);
            ua.u[1] = *(const uint4*)(zr1 + kb + 16);
            a1[kt] = ua.v;
        }
    }

    float best0[8], best1[8];
    int   bidx0[8], bidx1[8];
#pragma unroll
    for (int r = 0; r < 8; ++r) {
        best0[r] = 3.402823466e38f; bidx0[r] = 0;
        best1[r] = 3.402823466e38f; bidx1[r] = 0;
    }

    // cooperative staging geometry: thread -> (row, 32B column chunk)
    const int srow = tid >> 4;            // 0..15
    const int scol = (tid & 15) * 16;     // 0..240

    // preload tile 'split' straight into LDS buffer 0
    {
        const uint4* src = (const uint4*)(cbf + (size_t)(split * 16 + srow) * D_CH + scol);
        uint4 x0 = src[0], x1 = src[1];
        *(uint4*)(&Bt[0][srow * BT_STRIDE + scol])     = x0;
        *(uint4*)(&Bt[0][srow * BT_STRIDE + scol + 8]) = x1;
    }
    __syncthreads();

    int cur = 0;
    for (int i = 0; i < NTILES / NSPLIT; ++i) {
        const int nt  = split + i * NSPLIT;
        int ntn = nt + NSPLIT;
        if (ntn >= NTILES) ntn = split;   // wrap: valid memory, values unused

        // issue next tile's global loads (latency hidden by WMMA burst below)
        const uint4* src = (const uint4*)(cbf + (size_t)(ntn * 16 + srow) * D_CH + scol);
        uint4 q0 = src[0], q1 = src[1];

        // ---- 2 x (16x16x256) via 16 WMMAs over two independent acc chains ----
        v8f acc0 = {}, acc1 = {};
        const ushort_t* brow = &Bt[cur][lo * BT_STRIDE];
#pragma unroll
        for (int kt = 0; kt < 8; ++kt) {
            int kb = kt * 32 + hi * 8;    // ISA 16-bit B layout
            union { uint4 u[2]; v16bf v; } ub;
            ub.u[0] = *(const uint4*)(brow + kb);
            ub.u[1] = *(const uint4*)(brow + kb + 16);
            acc0 = __builtin_amdgcn_wmma_f32_16x16x32_bf16(false, a0[kt], false, ub.v,
                                                           (short)0, acc0, false, false);
            acc1 = __builtin_amdgcn_wmma_f32_16x16x32_bf16(false, a1[kt], false, ub.v,
                                                           (short)0, acc1, false, false);
        }

        // store prefetched tile into the other buffer while WMMAs drain
        *(uint4*)(&Bt[cur ^ 1][srow * BT_STRIDE + scol])     = q0;
        *(uint4*)(&Bt[cur ^ 1][srow * BT_STRIDE + scol + 8]) = q1;

        // d = ||e||^2 - 2<z,e>; C layout: acc[r] -> (M = r + 8*hi, N = lo)
        const int n = nt * 16 + lo;
        const float sq = sumsq[n];
#pragma unroll
        for (int r = 0; r < 8; ++r) {
            float d0 = fmaf(-2.0f, acc0[r], sq);
            float d1 = fmaf(-2.0f, acc1[r], sq);
            if (d0 < best0[r]) { best0[r] = d0; bidx0[r] = n; }   // n ascending: first min kept
            if (d1 < best1[r]) { best1[r] = d1; bidx1[r] = n; }
        }

        __syncthreads();   // next-buffer stores visible; current buffer free for overwrite
        cur ^= 1;
    }

    // reduce across the 16 lanes sharing each M (first-index tie-break)
#pragma unroll
    for (int off = 1; off < 16; off <<= 1) {
#pragma unroll
        for (int r = 0; r < 8; ++r) {
            float ob = __shfl_xor(best0[r], off, 32);
            int   oi = __shfl_xor(bidx0[r], off, 32);
            if (ob < best0[r] || (ob == best0[r] && oi < bidx0[r])) { best0[r] = ob; bidx0[r] = oi; }
            ob = __shfl_xor(best1[r], off, 32);
            oi = __shfl_xor(bidx1[r], off, 32);
            if (ob < best1[r] || (ob == best1[r] && oi < bidx1[r])) { best1[r] = ob; bidx1[r] = oi; }
        }
    }
    if (lo == 0) {
#pragma unroll
        for (int r = 0; r < 8; ++r) {
            int m = m0 + hi * 8 + r;
            bestd_ws[split * NTOK + m] = best0[r];
            bestn_ws[split * NTOK + m] = bidx0[r];
            bestd_ws[split * NTOK + m + 16] = best1[r];
            bestn_ws[split * NTOK + m + 16] = bidx1[r];
        }
    }
}

// ---------------- kernel 5: combine N-splits, emit index + presence ----------------
__global__ __launch_bounds__(256) void vq_combine(const float* __restrict__ bestd_ws,
                                                  const int*   __restrict__ bestn_ws,
                                                  int*   __restrict__ idxbuf,
                                                  float* __restrict__ out_idx,
                                                  unsigned* __restrict__ presence) {
    int t = blockIdx.x * 256 + threadIdx.x;        // 64 blocks -> 16384
    float bd = bestd_ws[t];
    int   bn = bestn_ws[t];
#pragma unroll
    for (int s = 1; s < NSPLIT; ++s) {
        float d2 = bestd_ws[s * NTOK + t];
        int   n2 = bestn_ws[s * NTOK + t];
        if (d2 < bd || (d2 == bd && n2 < bn)) { bd = d2; bn = n2; }
    }
    idxbuf[t]  = bn;
    out_idx[t] = (float)bn;
    presence[(t >> 10) * NE + bn] = 1u;            // benign same-value race
}

// ---------------- kernel 6: gather z_q (channel-first) + loss partial sums ----------------
__global__ __launch_bounds__(256) void vq_gather(const float* __restrict__ z,
                                                 const float* __restrict__ cb,
                                                 const int*   __restrict__ idxbuf,
                                                 float* __restrict__ zq_out,
                                                 float* __restrict__ blocksums) {
    int gid = blockIdx.x * 256 + threadIdx.x;      // linear over [B,D,H,W]
    int hw = gid & 1023;
    int bd = gid >> 10;
    int d  = bd & 255;
    int b  = bd >> 8;
    int t  = b * HW + hw;
    int n  = idxbuf[t];
    float c  = cb[(size_t)n * D_CH + d];           // exact f32 codebook value
    float zv = z[gid];                             // channel-first index == output index
    zq_out[gid] = c;
    float diff = c - zv;
    __shared__ float red[256];
    red[threadIdx.x] = diff * diff;
    __syncthreads();
#pragma unroll
    for (int s = 128; s > 0; s >>= 1) {
        if (threadIdx.x < s) red[threadIdx.x] += red[threadIdx.x + s];
        __syncthreads();
    }
    if (threadIdx.x == 0) blocksums[blockIdx.x] = red[0];
}

// ---------------- kernel 7: deterministic finalize (loss + diversity) ----------------
__global__ __launch_bounds__(256) void vq_finalize(const float* __restrict__ blocksums,
                                                   const unsigned* __restrict__ presence,
                                                   float* __restrict__ out_loss,
                                                   float* __restrict__ out_div) {
    __shared__ float    redf[256];
    __shared__ unsigned redu[256];
    float s = 0.f;
    for (int i = threadIdx.x; i < GATHER_BLOCKS; i += 256) s += blocksums[i];
    unsigned c = 0u;
    for (int i = threadIdx.x; i < B_SZ * NE; i += 256) c += presence[i];
    redf[threadIdx.x] = s;
    redu[threadIdx.x] = c;
    __syncthreads();
#pragma unroll
    for (int st = 128; st > 0; st >>= 1) {
        if (threadIdx.x < st) {
            redf[threadIdx.x] += redf[threadIdx.x + st];
            redu[threadIdx.x] += redu[threadIdx.x + st];
        }
        __syncthreads();
    }
    if (threadIdx.x == 0) {
        // loss = (1 + BETA) * mean(diff^2), BETA = 0.25
        out_loss[0] = 1.25f * redf[0] / (float)ZQ_ELEMS;
        out_div[0]  = (float)redu[0] / (float)NTOK;
    }
}

// ---------------- launcher ----------------
extern "C" void kernel_launch(void* const* d_in, const int* in_sizes, int n_in,
                              void* d_out, int out_size, void* d_ws, size_t ws_size,
                              hipStream_t stream) {
    const float* z  = (const float*)d_in[0];   // [16,256,32,32]
    const float* cb = (const float*)d_in[1];   // [8192,256]
    float* out      = (float*)d_out;
    float* out_zq   = out;
    float* out_idx  = out + OUT_IDX_OFF;
    float* out_loss = out + OUT_LOSS_OFF;
    float* out_div  = out + OUT_DIV_OFF;

    // zt (bf16, 8 MiB) staged inside the 16 MiB z_q output region (consumed before gather)
    ushort_t* zt = (ushort_t*)out_zq;

    char* ws = (char*)d_ws;                    // ~5.7 MiB total
    ushort_t* cbf      = (ushort_t*)(ws);                       // 4,194,304 B
    float*    sumsq    = (float*)   (ws + 4194304);             //    32,768 B
    int*      idxbuf   = (int*)     (ws + 4227072);             //    65,536 B
    unsigned* presence = (unsigned*)(ws + 4292608);             //   524,288 B
    float*    blocksum = (float*)   (ws + 4816896);             //    65,536 B
    float*    bestd    = (float*)   (ws + 4882432);             //   524,288 B (8 x 16384)
    int*      bestn    = (int*)     (ws + 5406720);             //   524,288 B

    vq_zero_presence<<<512,  256, 0, stream>>>(presence);
    vq_prep_z       <<<64,   256, 0, stream>>>(z, zt);
    vq_prep_cb      <<<1024, 256, 0, stream>>>(cb, cbf, sumsq);
    vq_argmin       <<<MBLOCKS * NSPLIT, 256, 0, stream>>>(zt, cbf, sumsq, bestd, bestn);
    vq_combine      <<<64,   256, 0, stream>>>(bestd, bestn, idxbuf, out_idx, presence);
    vq_gather       <<<GATHER_BLOCKS, 256, 0, stream>>>(z, cb, idxbuf, out_zq, blocksum);
    vq_finalize     <<<1,    256, 0, stream>>>(blocksum, presence, out_loss, out_div);
}
